// _SupportBiasedAttention_84439057039406
// MI455X (gfx1250) — matrix operand order repaired and checked by reading, not compiled, not run on hardware
//
#include <hip/hip_runtime.h>

// ---------------------------------------------------------------------------
// SupportBiasedAttention for MI455X (gfx1250), wave32 + WMMA bf16 + TDM.
// B=4, Q=1024, K=2048, QDIM=MDIM=EDIM=1024, H=16, HD=64.
// Pipeline:
//   1) convert query/memory to bf16; transpose+convert Wq/Wk/Wv/Wo to bf16
//   2) TDM-fed WMMA GEMMs -> Qp[B,H,Q,64], Kp[B,H,K,64], VpT[B,H,64,K]
//   3) fused flash attention: WG = (b,h,128 queries), K/V tiles shared in LDS
//   4) TDM-fed WMMA GEMM attn @ WoT -> f32 output
// ---------------------------------------------------------------------------

typedef __attribute__((ext_vector_type(8)))  __bf16 v8bf;
typedef __attribute__((ext_vector_type(16))) __bf16 v16bf;
typedef __attribute__((ext_vector_type(8)))  float  v8f;
typedef unsigned int u32x4 __attribute__((ext_vector_type(4)));
typedef int i32x8 __attribute__((ext_vector_type(8)));
typedef int i32x4 __attribute__((ext_vector_type(4)));

#define HEADS 16
#define HDIM  64
#define QLEN  1024
#define KLEN  2048
#define EDIM  1024
#define PSTE  72        // padded LDS row stride in bf16 elements (144 bytes)

__device__ __forceinline__ v16bf frag_ld(const __bf16* lo, const __bf16* hi) {
  v8bf a = *(const v8bf*)lo;
  v8bf b = *(const v8bf*)hi;
  v16bf f;
#pragma unroll
  for (int e = 0; e < 8; ++e) { f[e] = a[e]; f[e + 8] = b[e]; }
  return f;
}

__device__ __forceinline__ v8f wmma_bf(v16bf a, v16bf b, v8f c) {
  return __builtin_amdgcn_wmma_f32_16x16x32_bf16(
      false, a, false, b, (short)0, c, false, false);
}

// ---------------------------------------------------------------------------
// TDM: load a 2D bf16 tile [tile_d1 rows x tile_d0 elems] from global to LDS.
// Row stride in global = d0_stride elems. LDS gets 16B padding every 128B
// stored -> LDS row stride = tile_d0*2 + 16 bytes (tile_d0 must be 64 here).
// Tensor dims == tile dims so the OOB logic never triggers.
// This toolchain's builtin is the 6-arg form:
//   (u32x4 g0, i32x8 g1, i32x4 g2, i32x4 g3, i32x8 unused, i32 cpol)
// ---------------------------------------------------------------------------
__device__ __forceinline__ void tdm_load_2d(unsigned lds_off, const void* gaddr,
                                            unsigned tile_d0, unsigned tile_d1,
                                            unsigned d0_stride) {
  unsigned long long ga = (unsigned long long)(uintptr_t)gaddr;
  u32x4 g0;
  g0[0] = 1u;                                   // count=1, user descriptor
  g0[1] = lds_off;                              // LDS byte address
  g0[2] = (unsigned)(ga & 0xffffffffu);         // global addr [31:0]
  g0[3] = (unsigned)((ga >> 32) & 0x01ffffffu) | (2u << 30);  // [56:32]|type=2
  i32x8 g1;
  // data_size=1 (2B) @[17:16]; pad_enable @20; pad_interval=4 (128B) @[24:22];
  // pad_amount=3 (4 dwords = 16B) @[31:25]; workgroup_mask=0
  g1[0] = (int)((1u << 16) | (1u << 20) | (4u << 22) | (3u << 25));
  g1[1] = (int)((tile_d0 & 0xffffu) << 16);                 // tensor_dim0 lo16
  g1[2] = (int)(((tile_d0 >> 16) & 0xffffu) | ((tile_d1 & 0xffffu) << 16));
  g1[3] = (int)(((tile_d1 >> 16) & 0xffffu) | ((tile_d0 & 0xffffu) << 16));
                                                // ^ tensor_dim1 hi16 | tile_dim0
  g1[4] = (int)(tile_d1 & 0xffffu);             // tile_dim1; tile_dim2=0
  g1[5] = (int)d0_stride;                       // tensor_dim0_stride [31:0]
  g1[6] = 0;                                    // stride hi + dim1_stride lo
  g1[7] = 0;
  i32x4 z4 = {0, 0, 0, 0};                      // groups 2/3: unused dims
  i32x8 z8 = {0, 0, 0, 0, 0, 0, 0, 0};          // trailing unused group
  __builtin_amdgcn_tensor_load_to_lds(g0, g1, z4, z4, z8, 0);
}

// ---------------------------------------------------------------------------
// Conversion kernels (bandwidth-trivial one-time prep).
// ---------------------------------------------------------------------------
__global__ __launch_bounds__(256) void cvt_bf16(const float* __restrict__ in,
                                                __bf16* __restrict__ out) {
  const size_t i = ((size_t)blockIdx.x * 256 + threadIdx.x) * 8;
  const float4 a = *(const float4*)(in + i);
  const float4 b = *(const float4*)(in + i + 4);
  v8bf o;
  o[0] = (__bf16)a.x; o[1] = (__bf16)a.y; o[2] = (__bf16)a.z; o[3] = (__bf16)a.w;
  o[4] = (__bf16)b.x; o[5] = (__bf16)b.y; o[6] = (__bf16)b.z; o[7] = (__bf16)b.w;
  *(v8bf*)(out + i) = o;
}

// W [N][N] f32 -> WT [n][k] bf16 (transpose), 32x32 LDS tiles.
__global__ __launch_bounds__(256) void wt_cvt(const float* __restrict__ W,
                                              __bf16* __restrict__ WT, int N) {
  __shared__ float t[32][33];
  const int tx = threadIdx.x & 31, ty = threadIdx.x >> 5;
  const int x0 = blockIdx.x * 32, y0 = blockIdx.y * 32;
#pragma unroll
  for (int p = 0; p < 4; ++p)
    t[ty + p * 8][tx] = W[(size_t)(y0 + ty + p * 8) * N + x0 + tx];
  __syncthreads();
#pragma unroll
  for (int p = 0; p < 4; ++p)
    WT[(size_t)(x0 + ty + p * 8) * N + y0 + tx] = (__bf16)t[tx][ty + p * 8];
}

// ---------------------------------------------------------------------------
// TDM-fed bf16 GEMM: C = A[M,Kd] * BT[N,Kd]^T.  Tile 128x128x64, 8 waves,
// wave = 32x64 region = 2x4 WMMA tiles x 2 k-steps. Double-buffered LDS.
// MODE 0: bf16 out, head-split [b,h,row,hd]   (Qp / Kp)
// MODE 1: bf16 out, V-transpose [b,h,hd,row]  (VpT)
// MODE 2: f32 out, row-major [row,col]        (final projection)
// ---------------------------------------------------------------------------
#define GT_A 0            // A tile at buf + 0        (128*144 = 18432 B)
#define GT_B 18432        // B tile at buf + 18432
#define GT_BUF 36864      // bytes per double-buffer slot

template <int MODE>
__global__ __launch_bounds__(256) void gemm_bf16(
    const __bf16* __restrict__ A, const __bf16* __restrict__ BT,
    __bf16* __restrict__ outb, float* __restrict__ outf,
    int M, int Kd, int N) {
  __shared__ __align__(16) char lds[2 * GT_BUF];
  const int tid  = threadIdx.x;
  const int lane = tid & 31, wave = tid >> 5;
  const int half = lane >> 4, idx = lane & 15;
  const int wm = wave >> 1, wn = wave & 1;
  const int b = blockIdx.z;
  const int rowBase = blockIdx.y * 128, colBase = blockIdx.x * 128;
  const __bf16* Ab = A + (size_t)b * M * Kd;
  const unsigned ldsu = (unsigned)(uintptr_t)&lds[0];
  const int nkt = Kd >> 6;

  v8f acc[2][4] = {};

  if (tid < 32) {
    tdm_load_2d(ldsu + GT_A, Ab + (size_t)rowBase * Kd, 64, 128, Kd);
    tdm_load_2d(ldsu + GT_B, BT + (size_t)colBase * Kd, 64, 128, Kd);
  }
  for (int kt = 0; kt < nkt; ++kt) {
    if (tid < 32) __builtin_amdgcn_s_wait_tensorcnt(0);
    __syncthreads();
    const int buf = kt & 1;
    if (kt + 1 < nkt && tid < 32) {
      const int nb = buf ^ 1;
      const int k0 = (kt + 1) << 6;
      tdm_load_2d(ldsu + nb * GT_BUF + GT_A, Ab + (size_t)rowBase * Kd + k0, 64, 128, Kd);
      tdm_load_2d(ldsu + nb * GT_BUF + GT_B, BT + (size_t)colBase * Kd + k0, 64, 128, Kd);
    }
    const __bf16* lA = (const __bf16*)&lds[buf * GT_BUF + GT_A];
    const __bf16* lB = (const __bf16*)&lds[buf * GT_BUF + GT_B];
#pragma unroll
    for (int step = 0; step < 2; ++step) {
      v16bf afr[2], bfr[4];
#pragma unroll
      for (int i = 0; i < 2; ++i) {
        const __bf16* p = lA + (wm * 32 + i * 16 + idx) * PSTE + step * 32;
        afr[i] = frag_ld(p + half * 8, p + 16 + half * 8);
      }
#pragma unroll
      for (int j = 0; j < 4; ++j) {
        const __bf16* p = lB + (wn * 64 + j * 16 + idx) * PSTE + step * 32;
        bfr[j] = frag_ld(p + half * 8, p + 16 + half * 8);
      }
#pragma unroll
      for (int i = 0; i < 2; ++i)
#pragma unroll
        for (int j = 0; j < 4; ++j)
          acc[i][j] = wmma_bf(afr[i], bfr[j], acc[i][j]);
    }
  }

#pragma unroll
  for (int i = 0; i < 2; ++i)
#pragma unroll
    for (int j = 0; j < 4; ++j)
#pragma unroll
      for (int v = 0; v < 8; ++v) {
        const int row = rowBase + wm * 32 + i * 16 + half * 8 + v;
        const int col = colBase + wn * 64 + j * 16 + idx;
        if (MODE == 0) {
          const int h = col >> 6, hd = col & 63;
          outb[(((size_t)b * HEADS + h) * M + row) * HDIM + hd] = (__bf16)acc[i][j][v];
        } else if (MODE == 1) {
          const int h = col >> 6, hd = col & 63;
          outb[(((size_t)b * HEADS + h) * HDIM + hd) * M + row] = (__bf16)acc[i][j][v];
        } else {
          outf[(size_t)row * N + col] = acc[i][j][v];
        }
      }
}

// ---------------------------------------------------------------------------
// Fused flash attention. WG = (b, h, 128 queries): 8 waves x 16 queries.
// K tile [64 keys][64 hd] and V tile [64 hd][64 keys] TDM-staged in LDS,
// double-buffered, shared by all 8 waves (8x less K/V traffic).
// ---------------------------------------------------------------------------
#define AT_K 0            // K tile in buffer        (64*144 = 9216 B)
#define AT_V 9216         // V tile in buffer
#define AT_BUF 18432      // bytes per double-buffer slot
#define AT_P  36864       // P staging base (per-wave 16*144 = 2304 B)

__global__ __launch_bounds__(256) void attn_flash(
    const __bf16* __restrict__ Qp, const __bf16* __restrict__ Kp,
    const __bf16* __restrict__ VpT, const float* __restrict__ bias,
    __bf16* __restrict__ attnO) {
  __shared__ __align__(16) char lds[AT_P + 8 * 2304];
  const int tid  = threadIdx.x;
  const int lane = tid & 31, wave = tid >> 5;
  const int half = lane >> 4, idx = lane & 15;
  const int qblk = blockIdx.x & 7;
  const int h    = (blockIdx.x >> 3) & 15;
  const int b    = blockIdx.x >> 7;
  const int qBase = qblk * 128 + wave * 16;

  const __bf16* qptr  = Qp  + (((size_t)b * HEADS + h) * QLEN + qBase) * HDIM;
  const __bf16* kbase = Kp  + ((size_t)b * HEADS + h) * KLEN * HDIM;
  const __bf16* vbase = VpT + ((size_t)b * HEADS + h) * HDIM * KLEN;
  const float*  bptr  = bias + (size_t)b * KLEN;
  const unsigned ldsu = (unsigned)(uintptr_t)&lds[0];
  __bf16* lpw = (__bf16*)&lds[AT_P + wave * 2304];

  // Q fragments: 16 rows x 64 hd = two 32-wide k-steps
  v16bf qa[2];
  {
    const __bf16* p = qptr + idx * HDIM;
    qa[0] = frag_ld(p + half * 8,      p + 16 + half * 8);
    qa[1] = frag_ld(p + 32 + half * 8, p + 48 + half * 8);
  }
  float mi[8], li[8];
#pragma unroll
  for (int v = 0; v < 8; ++v) { mi[v] = -3.0e38f; li[v] = 0.0f; }
  v8f o[4] = {};
  const float scale = 0.125f;
  const int nc = KLEN / 64;   // 32 chunks

  if (tid < 32) {
    tdm_load_2d(ldsu + AT_K, kbase, 64, 64, HDIM);
    tdm_load_2d(ldsu + AT_V, vbase, 64, 64, KLEN);
  }
  for (int c = 0; c < nc; ++c) {
    if (tid < 32) __builtin_amdgcn_s_wait_tensorcnt(0);
    __syncthreads();
    const int buf = c & 1;
    if (c + 1 < nc && tid < 32) {
      const int nb = buf ^ 1, kc1 = (c + 1) * 64;
      tdm_load_2d(ldsu + nb * AT_BUF + AT_K, kbase + (size_t)kc1 * HDIM, 64, 64, HDIM);
      tdm_load_2d(ldsu + nb * AT_BUF + AT_V, vbase + kc1,                64, 64, KLEN);
    }
    const __bf16* lK = (const __bf16*)&lds[buf * AT_BUF + AT_K];
    const __bf16* lV = (const __bf16*)&lds[buf * AT_BUF + AT_V];
    const int kc = c * 64;

    // ---- S = Q*K^T : 4 n-tiles of 16 keys x 2 hd k-steps = 8 WMMAs ----
    v8f s[4] = {};
#pragma unroll
    for (int t = 0; t < 4; ++t) {
      const __bf16* kr = lK + (t * 16 + idx) * PSTE;
      s[t] = wmma_bf(qa[0], frag_ld(kr + half * 8,      kr + 16 + half * 8),      s[t]);
      s[t] = wmma_bf(qa[1], frag_ld(kr + 32 + half * 8, kr + 48 + half * 8), s[t]);
    }
    float bv[4];
#pragma unroll
    for (int t = 0; t < 4; ++t) bv[t] = bptr[kc + t * 16 + idx];

    // ---- online softmax over 64 keys; row (v+8*half) spans 16 lanes ----
#pragma unroll
    for (int v = 0; v < 8; ++v) {
      float tv[4];
      float mx = -3.0e38f;
#pragma unroll
      for (int t = 0; t < 4; ++t) { tv[t] = s[t][v] * scale + bv[t]; mx = fmaxf(mx, tv[t]); }
#pragma unroll
      for (int d = 1; d < 16; d <<= 1) mx = fmaxf(mx, __shfl_xor(mx, d, 32));
      const float mnew = fmaxf(mi[v], mx);
      const float sc = __expf(mi[v] - mnew);
      mi[v] = mnew;
      float rs = 0.0f;
      const int r = v + 8 * half;
#pragma unroll
      for (int t = 0; t < 4; ++t) {
        const float pt = __expf(tv[t] - mnew);
        rs += pt;
        lpw[r * PSTE + t * 16 + idx] = (__bf16)pt;
      }
#pragma unroll
      for (int d = 1; d < 16; d <<= 1) rs += __shfl_xor(rs, d, 32);
      li[v] = li[v] * sc + rs;
#pragma unroll
      for (int j = 0; j < 4; ++j) o[j][v] *= sc;
    }
    asm volatile("s_wait_dscnt 0x0" ::: "memory");

    // ---- O += P*V : reload P as A-frags, 4 hd n-tiles x 2 k-steps ----
    const __bf16* pp = lpw + idx * PSTE;
    const v16bf pa0 = frag_ld(pp + half * 8,      pp + 16 + half * 8);
    const v16bf pa1 = frag_ld(pp + 32 + half * 8, pp + 48 + half * 8);
#pragma unroll
    for (int j = 0; j < 4; ++j) {
      const __bf16* vr = lV + (j * 16 + idx) * PSTE;
      o[j] = wmma_bf(pa0, frag_ld(vr + half * 8,      vr + 16 + half * 8),      o[j]);
      o[j] = wmma_bf(pa1, frag_ld(vr + 32 + half * 8, vr + 48 + half * 8), o[j]);
    }
  }

  // normalize + scatter into [B,Q,EDIM] bf16
#pragma unroll
  for (int v = 0; v < 8; ++v) {
    const float inv = 1.0f / li[v];
    const int row = qBase + v + 8 * half;
    const size_t base = ((size_t)b * QLEN + row) * EDIM + h * 64;
#pragma unroll
    for (int j = 0; j < 4; ++j)
      attnO[base + j * 16 + idx] = (__bf16)(o[j][v] * inv);
  }
}

// ---------------------------------------------------------------------------
// Workspace (72 MiB):
//  [ 0Mi) qbf  bf16 [B,Q,1024]   8Mi   (aliased later by attn output, same size)
//  [ 8Mi) mbf  bf16 [B,K,1024]  16Mi
//  [24Mi) WqT  [26Mi) WkT  [28Mi) WvT  [30Mi) WoT   bf16 [1024][1024], 2Mi each
//  [32Mi) Qp   bf16 [B,H,Q,64]   8Mi
//  [40Mi) Kp   bf16 [B,H,K,64]  16Mi
//  [56Mi) VpT  bf16 [B,H,64,K]  16Mi
// ---------------------------------------------------------------------------
extern "C" void kernel_launch(void* const* d_in, const int* in_sizes, int n_in,
                              void* d_out, int out_size, void* d_ws, size_t ws_size,
                              hipStream_t stream) {
  (void)in_sizes; (void)n_in; (void)out_size; (void)ws_size;
  const float* query  = (const float*)d_in[0];
  const float* memory = (const float*)d_in[1];
  const float* bias   = (const float*)d_in[2];
  const float* Wq     = (const float*)d_in[3];
  const float* Wk     = (const float*)d_in[4];
  const float* Wv     = (const float*)d_in[5];
  const float* Wo     = (const float*)d_in[6];
  float* out = (float*)d_out;

  char* ws = (char*)d_ws;
  const size_t Mi = (size_t)1 << 20;
  __bf16* qbf  = (__bf16*)(ws);
  __bf16* mbf  = (__bf16*)(ws + 8  * Mi);
  __bf16* WqT  = (__bf16*)(ws + 24 * Mi);
  __bf16* WkT  = (__bf16*)(ws + 26 * Mi);
  __bf16* WvT  = (__bf16*)(ws + 28 * Mi);
  __bf16* WoT  = (__bf16*)(ws + 30 * Mi);
  __bf16* Qp   = (__bf16*)(ws + 32 * Mi);
  __bf16* Kp   = (__bf16*)(ws + 40 * Mi);
  __bf16* VpT  = (__bf16*)(ws + 56 * Mi);
  __bf16* attn = (__bf16*)(ws);            // aliases qbf (dead by then)

  dim3 blk(256);
  cvt_bf16<<<dim3(2048), blk, 0, stream>>>(query,  qbf);   // 4M elems
  cvt_bf16<<<dim3(4096), blk, 0, stream>>>(memory, mbf);   // 8M elems
  wt_cvt<<<dim3(32, 32), blk, 0, stream>>>(Wq, WqT, 1024);
  wt_cvt<<<dim3(32, 32), blk, 0, stream>>>(Wk, WkT, 1024);
  wt_cvt<<<dim3(32, 32), blk, 0, stream>>>(Wv, WvT, 1024);
  wt_cvt<<<dim3(32, 32), blk, 0, stream>>>(Wo, WoT, 1024);

  gemm_bf16<0><<<dim3(8,  8, 4), blk, 0, stream>>>(qbf, WqT, Qp,  nullptr, QLEN, 1024, EDIM);
  gemm_bf16<0><<<dim3(8, 16, 4), blk, 0, stream>>>(mbf, WkT, Kp,  nullptr, KLEN, 1024, EDIM);
  gemm_bf16<1><<<dim3(8, 16, 4), blk, 0, stream>>>(mbf, WvT, VpT, nullptr, KLEN, 1024, EDIM);

  attn_flash<<<dim3(512), blk, 0, stream>>>(Qp, Kp, VpT, bias, attn);

  gemm_bf16<2><<<dim3(8, 32, 1), blk, 0, stream>>>(attn, WoT, nullptr, out, 4 * QLEN, 1024, EDIM);
}
